// MEMOIRAdapter_4922032521693
// MI455X (gfx1250) — matrix operand order repaired and checked by reading, not compiled, not run on hardware
//
#include <hip/hip_runtime.h>
#include <hip/hip_bf16.h>

// ---------------------------------------------------------------------------
// MEMOIRAdapter.new_weight_forward for MI455X (gfx1250, wave32, WMMA)
//
// Pipeline:
//   K1: top-2048 of |x[0, pb, :]| (exact rank select, jax tie-break)
//       -> active byte mask via permutation scatter            (1 block)
//   K2: overlap counts vs 64 masks, argmax, gate>=0.3 folded
//       into per-d float scale mscale[d] in {0,1}              (1 block)
//   K3: out[m,o] = sum_d (x[m,d]*mscale[d]) * W[o,d]
//       4096x4096x4096 f32 GEMM via split-precision f16 WMMA:
//       a = ah + al (fp16 hi/lo), C += Ah*Bh + Ah*Bl + Al*Bh
//       using v_wmma_f32_16x16x32_f16 with f32 accumulation.
// ---------------------------------------------------------------------------

typedef __attribute__((ext_vector_type(16))) _Float16 v16h;
typedef __attribute__((ext_vector_type(8)))  _Float16 v8h;
typedef __attribute__((ext_vector_type(4)))  _Float16 v4h;
typedef __attribute__((ext_vector_type(8)))  float    v8f;

#define TOPK_K      2048
#define IRR_THRESH  0.3f
#define DMAX        4096

#define BM   128
#define BN   128
#define BK   32
#define LDT  40      // padded LDS row length in f16 elements (80B: conflict-free)

// ---------------------------------------------------------------------------
// Kernel 1: exact top-K selection + permutation scatter into active mask
// rank(i) = #{j : a_j > a_i} + #{j < i : a_j == a_i}  (matches lax.top_k ties)
// ---------------------------------------------------------------------------
__global__ void memoir_topk_mask(const float* __restrict__ x,
                                 const int*   __restrict__ perm,
                                 const int*   __restrict__ pb_ptr,
                                 unsigned char* __restrict__ active,
                                 int D)
{
    __shared__ float absf[DMAX];
    const int pb = pb_ptr[0];
    const float* f0 = x + (size_t)pb * D;   // batch 0, row pb

    for (int i = threadIdx.x; i < D; i += blockDim.x) {
        absf[i]   = fabsf(f0[i]);
        active[i] = 0;
    }
    __syncthreads();

    for (int i = threadIdx.x; i < D; i += blockDim.x) {
        const float v = absf[i];
        int rank = 0;
        for (int j = 0; j < D; ++j) {
            const float u = absf[j];
            rank += (u > v) || ((u == v) && (j < i));
        }
        if (rank < TOPK_K) {
            active[perm[i]] = 1;
        }
    }
}

// ---------------------------------------------------------------------------
// Kernel 2: overlap counts -> best mask -> gated float scale per feature
// ---------------------------------------------------------------------------
__global__ void memoir_select_mask(const unsigned char* __restrict__ masks,
                                   const unsigned char* __restrict__ active,
                                   float* __restrict__ mscale,
                                   int D, int NM)
{
    __shared__ int   counts[64];
    __shared__ int   bestIdx;
    __shared__ float gate;

    const int tid = threadIdx.x;
    if (tid < NM) {
        const unsigned char* mrow = masks + (size_t)tid * D;
        int c = 0;
        for (int d = 0; d < D; ++d)
            c += (active[d] && mrow[d]) ? 1 : 0;
        counts[tid] = c;
    }
    __syncthreads();

    if (tid == 0) {
        int b = 0, bc = counts[0];
        for (int m = 1; m < NM; ++m)
            if (counts[m] > bc) { bc = counts[m]; b = m; }   // first-max (argmax)
        bestIdx = b;
        gate = (((float)bc / (float)TOPK_K) >= IRR_THRESH) ? 1.0f : 0.0f;
    }
    __syncthreads();

    const unsigned char* best = masks + (size_t)bestIdx * D;
    const float g = gate;
    for (int d = tid; d < D; d += blockDim.x)
        mscale[d] = g * (best[d] ? 1.0f : 0.0f);
}

// ---------------------------------------------------------------------------
// Kernel 3: split-precision f16 WMMA GEMM
//   out[m,n] = sum_k (X[m,k]*mscale[k]) * W[n,k],  M=N=K=4096
// Block: 256 threads (8 waves). Block tile 128x128x32. Wave tile 64x32.
// ---------------------------------------------------------------------------
__global__ __launch_bounds__(256)
void memoir_gemm_splitf16(const float* __restrict__ X,
                          const float* __restrict__ W,
                          const float* __restrict__ mscale,
                          float* __restrict__ out,
                          int M, int N, int K)
{
    __shared__ _Float16 Ah[BM * LDT];
    __shared__ _Float16 Al[BM * LDT];
    __shared__ _Float16 Bh[BM * LDT];
    __shared__ _Float16 Bl[BM * LDT];

    const int tid      = threadIdx.x;
    const int lane     = tid & 31;
    const int wave     = tid >> 5;
    const int wm       = wave & 1;      // 2 waves along M
    const int wn       = wave >> 1;     // 4 waves along N
    const int lane15   = lane & 15;
    const int laneHalf = lane >> 4;     // 0/1

    const int mBase = blockIdx.y * BM;
    const int nBase = blockIdx.x * BN;

    v8f acc[4][2];
    #pragma unroll
    for (int ms = 0; ms < 4; ++ms)
        #pragma unroll
        for (int ns = 0; ns < 2; ++ns)
            acc[ms][ns] = (v8f)(0.0f);

    union Frag { v16h v; v8h p[2]; };

    for (int k0 = 0; k0 < K; k0 += BK) {
        // ---- global -> LDS, f32 masked then split into f16 hi/lo ----------
        #pragma unroll
        for (int i = 0; i < 4; ++i) {
            const int f4  = tid + 256 * i;   // float4 id, 0..1023
            const int row = f4 >> 3;         // 0..127
            const int c4  = (f4 & 7) * 4;    // 0,4,...,28

            const float4 ms4 = *reinterpret_cast<const float4*>(mscale + k0 + c4);
            const float4 av  = *reinterpret_cast<const float4*>(
                                   X + (size_t)(mBase + row) * K + k0 + c4);
            const float4 bv  = *reinterpret_cast<const float4*>(
                                   W + (size_t)(nBase + row) * K + k0 + c4);

            const float a0 = av.x * ms4.x, a1 = av.y * ms4.y,
                        a2 = av.z * ms4.z, a3 = av.w * ms4.w;

            v4h ahi, alo, bhi, blo;
            ahi.x = (_Float16)a0;   alo.x = (_Float16)(a0 - (float)ahi.x);
            ahi.y = (_Float16)a1;   alo.y = (_Float16)(a1 - (float)ahi.y);
            ahi.z = (_Float16)a2;   alo.z = (_Float16)(a2 - (float)ahi.z);
            ahi.w = (_Float16)a3;   alo.w = (_Float16)(a3 - (float)ahi.w);
            bhi.x = (_Float16)bv.x; blo.x = (_Float16)(bv.x - (float)bhi.x);
            bhi.y = (_Float16)bv.y; blo.y = (_Float16)(bv.y - (float)bhi.y);
            bhi.z = (_Float16)bv.z; blo.z = (_Float16)(bv.z - (float)bhi.z);
            bhi.w = (_Float16)bv.w; blo.w = (_Float16)(bv.w - (float)bhi.w);

            *reinterpret_cast<v4h*>(&Ah[row * LDT + c4]) = ahi;
            *reinterpret_cast<v4h*>(&Al[row * LDT + c4]) = alo;
            *reinterpret_cast<v4h*>(&Bh[row * LDT + c4]) = bhi;
            *reinterpret_cast<v4h*>(&Bl[row * LDT + c4]) = blo;
        }
        __syncthreads();

        // ---- prefetch next k-tile (emits global_prefetch_b8) --------------
        if (k0 + BK < K) {
            const int prow  = tid & 127;
            const int ppart = tid >> 7;   // 0/1 -> two 64B halves of the row
            __builtin_prefetch(X + (size_t)(mBase + prow) * K + (k0 + BK) + ppart * 16, 0, 0);
            __builtin_prefetch(W + (size_t)(nBase + prow) * K + (k0 + BK) + ppart * 16, 0, 0);
        }

        // ---- B fragments: 32x16 (KxN), lane(0-15)=N,K0-15; lane(16-31)=K16-31
        Frag bhF[2], blF[2];
        #pragma unroll
        for (int ns = 0; ns < 2; ++ns) {
            const int row = wn * 32 + ns * 16 + lane15;  // N index inside tile
            const int kc  = laneHalf * 16;               // K offset
            bhF[ns].p[0] = *reinterpret_cast<const v8h*>(&Bh[row * LDT + kc]);
            bhF[ns].p[1] = *reinterpret_cast<const v8h*>(&Bh[row * LDT + kc + 8]);
            blF[ns].p[0] = *reinterpret_cast<const v8h*>(&Bl[row * LDT + kc]);
            blF[ns].p[1] = *reinterpret_cast<const v8h*>(&Bl[row * LDT + kc + 8]);
        }

        // ---- A fragments + 3 WMMAs per 16x16 subtile ----------------------
        #pragma unroll
        for (int ms = 0; ms < 4; ++ms) {
            const int row = wm * 64 + ms * 16 + lane15;  // M index inside tile
            const int c0  = laneHalf * 16;               // K 0-7 / 16-23
            const int c1  = laneHalf * 16 + 8;           // K 8-15 / 24-31
            Frag ahF, alF;
            ahF.p[0] = *reinterpret_cast<const v8h*>(&Ah[row * LDT + c0]);
            ahF.p[1] = *reinterpret_cast<const v8h*>(&Ah[row * LDT + c1]);
            alF.p[0] = *reinterpret_cast<const v8h*>(&Al[row * LDT + c0]);
            alF.p[1] = *reinterpret_cast<const v8h*>(&Al[row * LDT + c1]);

            #pragma unroll
            for (int ns = 0; ns < 2; ++ns) {
                acc[ms][ns] = __builtin_amdgcn_wmma_f32_16x16x32_f16(
                    false, ahF.v, false, bhF[ns].v, (short)0, acc[ms][ns], false, false);
                acc[ms][ns] = __builtin_amdgcn_wmma_f32_16x16x32_f16(
                    false, ahF.v, false, blF[ns].v, (short)0, acc[ms][ns], false, false);
                acc[ms][ns] = __builtin_amdgcn_wmma_f32_16x16x32_f16(
                    false, alF.v, false, bhF[ns].v, (short)0, acc[ms][ns], false, false);
            }
        }
        __syncthreads();
    }

    // ---- epilogue: C/D layout: lane(0-15): M=r,N=lane ; lane(16-31): M=8+r
    #pragma unroll
    for (int ms = 0; ms < 4; ++ms) {
        #pragma unroll
        for (int ns = 0; ns < 2; ++ns) {
            const int col = nBase + wn * 32 + ns * 16 + lane15;
            #pragma unroll
            for (int r = 0; r < 8; ++r) {
                const int row = mBase + wm * 64 + ms * 16 + laneHalf * 8 + r;
                out[(size_t)row * N + col] = acc[ms][ns][r];
            }
        }
    }
}

// ---------------------------------------------------------------------------
extern "C" void kernel_launch(void* const* d_in, const int* in_sizes, int n_in,
                              void* d_out, int out_size, void* d_ws, size_t ws_size,
                              hipStream_t stream)
{
    (void)n_in; (void)out_size; (void)ws_size;

    const float*         x     = (const float*)d_in[0];          // (B,S,D) f32
    const float*         w     = (const float*)d_in[1];          // (O,D)   f32
    const unsigned char* masks = (const unsigned char*)d_in[2];  // (64,D)  bool
    const int*           perm  = (const int*)d_in[3];            // (D,)    int
    const int*           pb    = (const int*)d_in[4];            // scalar  int

    const int D  = in_sizes[3];          // 4096
    const int O  = in_sizes[1] / D;      // 4096
    const int NM = in_sizes[2] / D;      // 64
    const int MS = in_sizes[0] / D;      // B*S = 4096

    unsigned char* active = (unsigned char*)d_ws;
    float*         mscale = (float*)((char*)d_ws + (((size_t)D + 255) & ~(size_t)255));

    memoir_topk_mask  <<<1, 256, 0, stream>>>(x, perm, pb, active, D);
    memoir_select_mask<<<1, 256, 0, stream>>>(masks, active, mscale, D, NM);

    dim3 grid(O / BN, MS / BM);   // 32 x 32 workgroups
    memoir_gemm_splitf16<<<grid, 256, 0, stream>>>(x, w, mscale, (float*)d_out,
                                                   MS, O, D);
}
